// ModConv_12876311953959
// MI455X (gfx1250) — compile-verified, matrix operand order checked by reading, not run on hardware
//
#include <hip/hip_runtime.h>

#define B_    8
#define C_IN  256
#define C_OUT 256
#define H_    128
#define W_    128
#define Z_    512
#define TAPS  9
#define EPS_  1e-8f

typedef __attribute__((ext_vector_type(16))) __bf16       v16bf;
typedef __attribute__((ext_vector_type(8)))  float        v8f;
typedef __attribute__((ext_vector_type(4)))  unsigned int v4u;

__device__ __forceinline__ unsigned short f2bf(float f) {
    unsigned u = __builtin_bit_cast(unsigned, f);
    u += 0x7FFFu + ((u >> 16) & 1u);           // round-to-nearest-even
    return (unsigned short)(u >> 16);
}

// ---------------- prep kernels ----------------

// style[b][i] = dot(w[b,:], affine_w[i,:]) + affine_b[i] + 1
__global__ void k_style(const float* __restrict__ w, const float* __restrict__ aw,
                        const float* __restrict__ ab, float* __restrict__ style) {
    int b = blockIdx.x, i = threadIdx.x;
    const float* wr = w + (size_t)b * Z_;
    const float* ar = aw + (size_t)i * Z_;
    float s = 0.f;
    for (int j = 0; j < Z_; ++j) s += wr[j] * ar[j];
    style[b * C_IN + i] = s + ab[i] + 1.0f;
}

// wsq[o][i] = sum over 9 taps of weight[o][i][kh][kw]^2
__global__ void k_wsq(const float* __restrict__ weight, float* __restrict__ wsq) {
    int o = blockIdx.x, i = threadIdx.x;
    const float* p = weight + ((size_t)o * C_IN + i) * TAPS;
    float s = 0.f;
    for (int t = 0; t < TAPS; ++t) { float v = p[t]; s += v * v; }
    wsq[o * C_IN + i] = s;
}

// demod[b][o] = rsqrt(sum_i wsq[o][i]*style[b][i]^2 + eps)
__global__ void k_demod(const float* __restrict__ wsq, const float* __restrict__ style,
                        float* __restrict__ demod) {
    int b = blockIdx.x, o = threadIdx.x;
    const float* sr = style + b * C_IN;
    const float* wr = wsq + (size_t)o * C_IN;
    float s = EPS_;
    for (int i = 0; i < C_IN; ++i) { float st = sr[i]; s += wr[i] * st * st; }
    demod[b * C_OUT + o] = rsqrtf(s);
}

// wbf[b][tap][o][i] = bf16(weight[o][i][kh][kw] * style[b][i] * demod[b][o])
__global__ void k_modw(const float* __restrict__ weight, const float* __restrict__ style,
                       const float* __restrict__ demod, unsigned short* __restrict__ wbf) {
    int o = blockIdx.x, t = blockIdx.y, b = blockIdx.z, i = threadIdx.x;
    float v = weight[((size_t)o * C_IN + i) * TAPS + t] * style[b * C_IN + i]
            * demod[b * C_OUT + o];
    wbf[(((size_t)(b * TAPS + t) * C_OUT) + o) * C_IN + i] = f2bf(v);
}

// ---------------- main implicit-GEMM conv ----------------
// One workgroup: sample b, 128 output channels (oblk), 2 output rows (y0, y0+1).
// 8 waves = 2 (M=64) x 4 (N=64); each wave owns a 64x64 tile = 4x4 WMMA frags.
// A (weights) fragments: loaded straight from global (WGP$ serves cross-wave reuse).
// B (activations): staged once per 32-channel chunk into LDS with halo + zero pad.

#define PXW       130   // 128 pixels + 1 halo each side
#define XS_STRIDE 40    // channel stride (halves); 80B, 16B aligned, conflict-friendly

__global__ void __launch_bounds__(256)
k_conv(const float* __restrict__ x, const unsigned short* __restrict__ wbf,
       float* __restrict__ out) {
    __shared__ __align__(16) unsigned short xsT[4 * PXW * XS_STRIDE];  // 41,600 B

    const int y0   = blockIdx.x * 2;  // output rows y0, y0+1
    const int oblk = blockIdx.y;      // which 128 of 256 output channels
    const int b    = blockIdx.z;
    const int tid  = threadIdx.x;
    const int lane = tid & 31;
    const int wid  = tid >> 5;        // 8 waves
    const int wm   = wid & 1;         // M block (64 rows)
    const int wn   = wid >> 1;        // N block (64 px)
    const int l16  = lane & 15;
    const int lhi  = (lane >= 16) ? 8 : 0;  // K-half select per ISA 16-bit A/B layout

    v8f acc[4][4];
    {
        v8f z = {0.f, 0.f, 0.f, 0.f, 0.f, 0.f, 0.f, 0.f};
        for (int fm = 0; fm < 4; ++fm)
            for (int fn = 0; fn < 4; ++fn) acc[fm][fn] = z;
    }

    for (int i0 = 0; i0 < C_IN; i0 += 32) {
        __syncthreads();  // previous chunk's compute done before overwriting xsT
        // ---- stage X: rows y0-1 .. y0+2, 32 channels, 130 padded cols, bf16 ----
        for (int e = tid; e < 4 * 16 * PXW; e += 256) {
            int yy   = e / (16 * PXW);
            int r1   = e % (16 * PXW);
            int ich2 = r1 / PXW;        // channel pair
            int px   = r1 % PXW;
            int row  = y0 + yy - 1;
            int col  = px - 1;
            float v0 = 0.f, v1 = 0.f;
            if (row >= 0 && row < H_ && col >= 0 && col < W_) {
                const float* xp = x + (((size_t)b * C_IN + i0 + 2 * ich2) * H_ + row) * W_ + col;
                v0 = xp[0];
                v1 = xp[(size_t)H_ * W_];
            }
            unsigned pk = (unsigned)f2bf(v0) | ((unsigned)f2bf(v1) << 16);
            *reinterpret_cast<unsigned*>(&xsT[(yy * PXW + px) * XS_STRIDE + 2 * ich2]) = pk;
        }
        __syncthreads();

        for (int t = 0; t < TAPS; ++t) {
            const int kh = t / 3, kw = t % 3;

            // A fragments straight from global modulated-weight buffer (bf16,
            // row-major [o][i]): per-lane two aligned 16B chunks match the ISA
            // 16-bit A 16x32 layout (K-halves split by lane half).
            v16bf afrag[4];
            for (int fm = 0; fm < 4; ++fm) {
                size_t rowbase =
                    (((size_t)(b * TAPS + t) * C_OUT) + oblk * 128 + wm * 64 + fm * 16 + l16)
                        * C_IN + i0 + lhi;
                union { v4u u[2]; v16bf v; } u;
                u.u[0] = *reinterpret_cast<const v4u*>(wbf + rowbase);
                u.u[1] = *reinterpret_cast<const v4u*>(wbf + rowbase + 16);
                afrag[fm] = u.v;
            }

            for (int fn = 0; fn < 4; ++fn) {
                int n   = wn * 64 + fn * 16 + l16;  // 0..255 over 2 output rows
                int r   = n >> 7;                   // which of the 2 rows
                int pxi = (n & 127) + kw;           // shifted padded column 0..129
                const unsigned short* bp = &xsT[((r + kh) * PXW + pxi) * XS_STRIDE];
                union { v4u u[2]; v16bf v; } u;
                u.u[0] = *reinterpret_cast<const v4u*>(&bp[lhi]);
                u.u[1] = *reinterpret_cast<const v4u*>(&bp[16 + lhi]);
                v16bf bfrag = u.v;

                for (int fm = 0; fm < 4; ++fm)
                    acc[fm][fn] = __builtin_amdgcn_wmma_f32_16x16x32_bf16(
                        false, afrag[fm], false, bfrag,
                        (short)0, acc[fm][fn], false, false);
            }
        }
    }

    // ---- epilogue: C/D layout — VGPR r holds M=r (lanes 0-15) / M=8+r (16-31) ----
    for (int fm = 0; fm < 4; ++fm) {
        for (int fn = 0; fn < 4; ++fn) {
            int n     = wn * 64 + fn * 16 + l16;
            int row   = y0 + (n >> 7);
            int px    = n & 127;
            int obase = oblk * 128 + wm * 64 + fm * 16 + lhi;
            for (int rr = 0; rr < 8; ++rr) {
                out[(((size_t)b * C_OUT + (obase + rr)) * H_ + row) * W_ + px] =
                    acc[fm][fn][rr];
            }
        }
    }
}

// ---------------- host ----------------

extern "C" void kernel_launch(void* const* d_in, const int* in_sizes, int n_in,
                              void* d_out, int out_size, void* d_ws, size_t ws_size,
                              hipStream_t stream) {
    const float* x      = (const float*)d_in[0];   // [8,256,128,128]
    const float* w      = (const float*)d_in[1];   // [8,512]
    const float* weight = (const float*)d_in[2];   // [256,256,3,3]
    const float* aw     = (const float*)d_in[3];   // [256,512]
    const float* ab     = (const float*)d_in[4];   // [256]
    float* out = (float*)d_out;                    // [8,256,128,128]

    char* ws = (char*)d_ws;
    float* style = (float*)(ws);                               // 8 KB
    float* demod = (float*)(ws + 8 * 1024);                    // 8 KB
    float* wsq   = (float*)(ws + 16 * 1024);                   // 256 KB
    unsigned short* wbf = (unsigned short*)(ws + 272 * 1024);  // 8*9*256*256 bf16 = 9.0 MB

    k_style<<<dim3(B_), dim3(C_IN), 0, stream>>>(w, aw, ab, style);
    k_wsq<<<dim3(C_OUT), dim3(C_IN), 0, stream>>>(weight, wsq);
    k_demod<<<dim3(B_), dim3(C_OUT), 0, stream>>>(wsq, style, demod);
    k_modw<<<dim3(C_OUT, TAPS, B_), dim3(C_IN), 0, stream>>>(weight, style, demod, wbf);
    k_conv<<<dim3(H_ / 2, 2, B_), dim3(256), 0, stream>>>(x, wbf, out);
}
